// TransformerBlock_81509889344261
// MI455X (gfx1250) — matrix operand order repaired and checked by reading, not compiled, not run on hardware
//
#include <hip/hip_runtime.h>
#include <hip/hip_bf16.h>
#include <math.h>
#include <stdint.h>

// ---------------------------------------------------------------------------
// CDNA5 (gfx1250) transformer block: LN -> QKV GEMM -> flash attention ->
// O-proj(+res) -> LN -> FC+GELU -> PR(+res).  All matmuls use
// v_wmma_f32_16x16x32_bf16 (wave32), fp32 accumulate, bf16 operands.
// GEMM is double-buffered; non-transposing tile stages use
// global_load_async_to_lds_b128 (ASYNCcnt) when the toolchain exposes it.
// ---------------------------------------------------------------------------

typedef __attribute__((ext_vector_type(16))) __bf16 v16bf;
typedef __attribute__((ext_vector_type(8)))  float  v8f;

union FragB { uint4 q[2]; v16bf v; };

#if defined(__gfx1250__) && __has_builtin(__builtin_amdgcn_global_load_async_to_lds_b128)
#define HAS_ASYNC_COPY 1
#else
#define HAS_ASYNC_COPY 0
#endif

#if HAS_ASYNC_COPY
// builtin signature (from compiler diagnostic): (int4 AS1*, int4 AS3*, imm, imm)
typedef int v4i_g __attribute__((vector_size(16)));
typedef __attribute__((address_space(1))) v4i_g as1_v4i;
typedef __attribute__((address_space(3))) v4i_g as3_v4i;
#endif

// copy 16 bytes (8 bf16) global -> LDS; async path writes LDS directly.
__device__ __forceinline__ void async_ld_b128(const unsigned short* g, unsigned short* l) {
#if HAS_ASYNC_COPY
  __builtin_amdgcn_global_load_async_to_lds_b128(
      (as1_v4i*)(uintptr_t)g, (as3_v4i*)(unsigned)(uintptr_t)l, 0, 0);
#else
  *(uint4*)l = *(const uint4*)g;
#endif
}

__device__ __forceinline__ void async_wait0() {
#if HAS_ASYNC_COPY
#if __has_builtin(__builtin_amdgcn_s_wait_asynccnt)
  __builtin_amdgcn_s_wait_asynccnt(0);
#else
  asm volatile("s_wait_asynccnt 0" ::: "memory");
#endif
#endif
}

__device__ __forceinline__ unsigned short f32_to_bf16(float f) {
  unsigned u = __float_as_uint(f);
  u += 0x7FFFu + ((u >> 16) & 1u);        // round-to-nearest-even
  return (unsigned short)(u >> 16);
}

__device__ __forceinline__ v8f zero8() {
  v8f z = {0.f,0.f,0.f,0.f,0.f,0.f,0.f,0.f};
  return z;
}

// --------------------------- fp32 -> bf16 convert ---------------------------
__global__ __launch_bounds__(256) void cvt_bf16_kernel(
    const float* __restrict__ src, unsigned short* __restrict__ dst, int n) {
  int i = blockIdx.x * blockDim.x + threadIdx.x;
  if (i < n) dst[i] = f32_to_bf16(src[i]);
}

// ------------------------------- LayerNorm ----------------------------------
__global__ __launch_bounds__(256) void ln_bf16_kernel(
    const float* __restrict__ x, const float* __restrict__ g,
    const float* __restrict__ b, unsigned short* __restrict__ y, int D) {
  int row = blockIdx.x;
  const float* xr = x + (size_t)row * D;
  float s = 0.f, s2 = 0.f;
  for (int i = threadIdx.x; i < D; i += 256) { float v = xr[i]; s += v; s2 += v * v; }
  for (int m = 1; m < 32; m <<= 1) { s += __shfl_xor(s, m, 32); s2 += __shfl_xor(s2, m, 32); }
  __shared__ float red0[8], red1[8];
  int wv = threadIdx.x >> 5, ln = threadIdx.x & 31;
  if (ln == 0) { red0[wv] = s; red1[wv] = s2; }
  __syncthreads();
  s = 0.f; s2 = 0.f;
  for (int w = 0; w < 8; w++) { s += red0[w]; s2 += red1[w]; }
  float mu  = s / (float)D;
  float var = s2 / (float)D - mu * mu;
  float inv = rsqrtf(var + 1e-5f);
  for (int i = threadIdx.x; i < D; i += 256) {
    float v = (xr[i] - mu) * inv * g[i] + b[i];
    y[(size_t)row * D + i] = f32_to_bf16(v);
  }
}

// ------------------------------ WMMA GEMM -----------------------------------
// C[M,N] = A[M,K](bf16) * B[K,N](bf16) + bias, opt exact GELU, opt fp32
// residual; fp32 and/or bf16 out.  Block tile 128x128, BK=32, double-buffered.
// 8 waves: 4(M) x 2(N); wave tile 32x64 -> 8 WMMA per K-step.
template<bool GELU, bool RES, bool OUTF, bool OUTB>
__global__ __launch_bounds__(256) void gemm_bf16_kernel(
    const unsigned short* __restrict__ A, const unsigned short* __restrict__ Bm,
    const float* __restrict__ bias, const float* __restrict__ res,
    float* __restrict__ outF, unsigned short* __restrict__ outB,
    int M, int N, int K) {
  __shared__ __align__(16) unsigned short As[2][128 * 32];   // [m][k]
  __shared__ __align__(16) unsigned short Bs[2][128 * 32];   // BT: [n][k]
  const int tid = threadIdx.x;
  const int wave = tid >> 5, lane = tid & 31;
  const int wm = wave & 3, wn = wave >> 2;
  const int r = lane & 15, hf = lane >> 4;
  const int m0 = blockIdx.x * 128, n0 = blockIdx.y * 128;
  const int rowA = tid >> 1, partA = tid & 1;   // A staging: 32B / thread
  const int kkB  = tid >> 3, npB  = tid & 7;    // B staging: 16 cols / thread

  v8f acc[2][4];
  for (int mi = 0; mi < 2; mi++)
    for (int nb = 0; nb < 4; nb++) acc[mi][nb] = zero8();

  const int nk = K / 32;

  // prologue: stage K-step 0 into buffer 0
  {
    const unsigned short* ga = A + (size_t)(m0 + rowA) * K + partA * 16;
    unsigned short* la = &As[0][rowA * 32 + partA * 16];
    async_ld_b128(ga, la);
    async_ld_b128(ga + 8, la + 8);
    const uint4* gb = (const uint4*)(Bm + (size_t)kkB * N + n0 + npB * 16);
    uint4 d0 = gb[0], d1 = gb[1];
    const unsigned short* e = (const unsigned short*)&d0;
    #pragma unroll
    for (int t = 0; t < 8; t++) Bs[0][(npB * 16 + t) * 32 + kkB] = e[t];
    e = (const unsigned short*)&d1;
    #pragma unroll
    for (int t = 0; t < 8; t++) Bs[0][(npB * 16 + 8 + t) * 32 + kkB] = e[t];
  }
  async_wait0();
  __syncthreads();

  for (int t = 0; t < nk; t++) {
    const int cur = t & 1, nxt = cur ^ 1;
    const bool have = (t + 1 < nk);
    uint4 d0 = {}, d1 = {};
    if (have) {   // issue next tile: A async -> LDS, B global -> regs
      const int k0n = (t + 1) * 32;
      const unsigned short* ga = A + (size_t)(m0 + rowA) * K + k0n + partA * 16;
      unsigned short* la = &As[nxt][rowA * 32 + partA * 16];
      async_ld_b128(ga, la);
      async_ld_b128(ga + 8, la + 8);
      const uint4* gb = (const uint4*)(Bm + (size_t)(k0n + kkB) * N + n0 + npB * 16);
      d0 = gb[0]; d1 = gb[1];
      if (t + 2 < nk)
        __builtin_prefetch(Bm + (size_t)((t + 2) * 32 + kkB) * N + n0 + npB * 16, 0, 0);
    }

    // compute on current buffer
    FragB af[2], bfr[4];
    #pragma unroll
    for (int mi = 0; mi < 2; mi++) {
      const unsigned short* base = &As[cur][(wm * 32 + mi * 16 + r) * 32];
      af[mi].q[0] = *(const uint4*)(base + hf * 8);          // K = h*8 + 0..7
      af[mi].q[1] = *(const uint4*)(base + 16 + hf * 8);     // K = 16 + h*8 + 0..7
    }
    #pragma unroll
    for (int nb = 0; nb < 4; nb++) {
      const unsigned short* base = &Bs[cur][(wn * 64 + nb * 16 + r) * 32];
      bfr[nb].q[0] = *(const uint4*)(base + hf * 8);         // K = h*8 + 0..7
      bfr[nb].q[1] = *(const uint4*)(base + 16 + hf * 8);    // K = 16 + h*8 + 0..7
    }
    #pragma unroll
    for (int mi = 0; mi < 2; mi++)
      #pragma unroll
      for (int nb = 0; nb < 4; nb++)
        acc[mi][nb] = __builtin_amdgcn_wmma_f32_16x16x32_bf16(
            false, af[mi].v, false, bfr[nb].v, (short)0, acc[mi][nb], false, false);

    if (have) {   // commit B (transposed) into next buffer
      const unsigned short* e = (const unsigned short*)&d0;
      #pragma unroll
      for (int q = 0; q < 8; q++) Bs[nxt][(npB * 16 + q) * 32 + kkB] = e[q];
      e = (const unsigned short*)&d1;
      #pragma unroll
      for (int q = 0; q < 8; q++) Bs[nxt][(npB * 16 + 8 + q) * 32 + kkB] = e[q];
    }
    async_wait0();
    __syncthreads();
  }

  // epilogue: C layout -> row M = h*8+i, col N = r
  #pragma unroll
  for (int mi = 0; mi < 2; mi++) {
    #pragma unroll
    for (int nb = 0; nb < 4; nb++) {
      int col = n0 + wn * 64 + nb * 16 + r;
      float bv = bias[col];
      #pragma unroll
      for (int i = 0; i < 8; i++) {
        int row = m0 + wm * 32 + mi * 16 + hf * 8 + i;
        float v = acc[mi][nb][i] + bv;
        if (GELU) v = 0.5f * v * (1.0f + erff(v * 0.70710678118654752f));
        if (RES)  v += res[(size_t)row * N + col];
        if (OUTF) outF[(size_t)row * N + col] = v;
        if (OUTB) outB[(size_t)row * N + col] = f32_to_bf16(v);
      }
    }
  }
}

// --------------------------- Flash attention --------------------------------
// qkv bf16 [B*S, 3D], q|k|v packed; head slice = cols head*64.. .
// grid: (S/128, H, B); 8 waves; wave owns 16 q-rows. HD = 64, causal.
__global__ __launch_bounds__(256) void attn_kernel(
    const unsigned short* __restrict__ qkv, unsigned short* __restrict__ out,
    int S, int Hn, int HD) {
  const int D3 = 3 * Hn * HD;          // 3072
  const int Dm = Hn * HD;              // 1024
  const int q0 = blockIdx.x * 128;
  const int head = blockIdx.y;
  const int bb = blockIdx.z;
  __shared__ __align__(16) unsigned short Qs[128 * 64];   // [q][hd]
  __shared__ __align__(16) unsigned short Ks[64 * 64];    // [key][hd] == BT for QK^T
  __shared__ __align__(16) unsigned short Vt[64 * 64];    // [hd][key] == BT for P.V
  __shared__ __align__(16) unsigned short Ps[8][16 * 64]; // per-wave P tile
  const int tid = threadIdx.x, wave = tid >> 5, lane = tid & 31;
  const int r = lane & 15, hf = lane >> 4;
  const size_t rowbase = (size_t)bb * S * D3;

  {   // stage Q tile (async, no transpose needed)
    int qr = tid >> 1, half = tid & 1;
    const unsigned short* g = qkv + rowbase + (size_t)(q0 + qr) * D3 + head * HD + half * 32;
    unsigned short* l = Qs + qr * 64 + half * 32;
    async_ld_b128(g, l);           async_ld_b128(g + 8, l + 8);
    async_ld_b128(g + 16, l + 16); async_ld_b128(g + 24, l + 24);
  }

  v8f o[4];
  for (int nb = 0; nb < 4; nb++) o[nb] = zero8();
  float mrow[8], lrow[8];
  for (int i = 0; i < 8; i++) { mrow[i] = -3.0e38f; lrow[i] = 0.f; }

  const int ktmax = (q0 + 127) / 64;
  for (int kt = 0; kt <= ktmax; kt++) {
    const int k0 = kt * 64;
    __syncthreads();
    {   // stage K [key][hd] (async) and V transposed [hd][key] (regs)
      int kr = tid >> 2, part = tid & 3;
      const unsigned short* gk = qkv + rowbase + (size_t)(k0 + kr) * D3 + Dm + head * HD + part * 16;
      unsigned short* lk = Ks + kr * 64 + part * 16;
      async_ld_b128(gk, lk);
      async_ld_b128(gk + 8, lk + 8);
      const uint4* vs = (const uint4*)(qkv + rowbase + (size_t)(k0 + kr) * D3 + 2 * Dm + head * HD + part * 16);
      uint4 v0 = vs[0], v1 = vs[1];
      const unsigned short* e = (const unsigned short*)&v0;
      #pragma unroll
      for (int t = 0; t < 8; t++) Vt[(part * 16 + t) * 64 + kr] = e[t];
      e = (const unsigned short*)&v1;
      #pragma unroll
      for (int t = 0; t < 8; t++) Vt[(part * 16 + 8 + t) * 64 + kr] = e[t];
    }
    async_wait0();   // covers Q on first iteration and this tile's K
    __syncthreads();

    // S = Q K^T (16 q-rows x 64 keys per wave)
    v8f s[4];
    for (int nb = 0; nb < 4; nb++) s[nb] = zero8();
    #pragma unroll
    for (int kk = 0; kk < 64; kk += 32) {
      FragB aq;
      const unsigned short* abase = Qs + (wave * 16 + r) * 64 + kk;
      aq.q[0] = *(const uint4*)(abase + hf * 8);
      aq.q[1] = *(const uint4*)(abase + 16 + hf * 8);
      #pragma unroll
      for (int nb = 0; nb < 4; nb++) {
        FragB bk;
        const unsigned short* bbase = Ks + (nb * 16 + r) * 64 + kk;
        bk.q[0] = *(const uint4*)(bbase + hf * 8);
        bk.q[1] = *(const uint4*)(bbase + 16 + hf * 8);
        s[nb] = __builtin_amdgcn_wmma_f32_16x16x32_bf16(
            false, aq.v, false, bk.v, (short)0, s[nb], false, false);
      }
    }

    // scale + causal mask + online softmax
    const float sc = 0.125f;   // 1/sqrt(64)
    #pragma unroll
    for (int i = 0; i < 8; i++) {
      int qg = q0 + wave * 16 + hf * 8 + i;
      float mx = -3.0e38f;
      #pragma unroll
      for (int nb = 0; nb < 4; nb++) {
        int kg = k0 + nb * 16 + r;
        float v = s[nb][i] * sc;
        if (kg > qg) v = -3.0e38f;
        s[nb][i] = v;
        mx = fmaxf(mx, v);
      }
      for (int m = 1; m < 16; m <<= 1) mx = fmaxf(mx, __shfl_xor(mx, m, 32));
      float mo = mrow[i];
      float mn = fmaxf(mo, mx);
      float corr = expf(mo - mn);
      mrow[i] = mn;
      float ps = 0.f;
      #pragma unroll
      for (int nb = 0; nb < 4; nb++) {
        float p = expf(s[nb][i] - mn);
        s[nb][i] = p;
        ps += p;
      }
      for (int m = 1; m < 16; m <<= 1) ps += __shfl_xor(ps, m, 32);
      lrow[i] = lrow[i] * corr + ps;
      #pragma unroll
      for (int nb = 0; nb < 4; nb++) {
        o[nb][i] *= corr;
        Ps[wave][(hf * 8 + i) * 64 + nb * 16 + r] = f32_to_bf16(s[nb][i]);
      }
    }
    __syncthreads();   // uniform; orders Ps write -> read

    // O += P V
    #pragma unroll
    for (int kk = 0; kk < 64; kk += 32) {
      FragB ap;
      const unsigned short* abase = Ps[wave] + r * 64 + kk;
      ap.q[0] = *(const uint4*)(abase + hf * 8);
      ap.q[1] = *(const uint4*)(abase + 16 + hf * 8);
      #pragma unroll
      for (int nb = 0; nb < 4; nb++) {
        FragB bv;
        const unsigned short* bbase = Vt + (nb * 16 + r) * 64 + kk;
        bv.q[0] = *(const uint4*)(bbase + hf * 8);
        bv.q[1] = *(const uint4*)(bbase + 16 + hf * 8);
        o[nb] = __builtin_amdgcn_wmma_f32_16x16x32_bf16(
            false, ap.v, false, bv.v, (short)0, o[nb], false, false);
      }
    }
  }

  // epilogue: O /= l, write bf16 [b*S+q, head*64 + n]
  #pragma unroll
  for (int nb = 0; nb < 4; nb++) {
    #pragma unroll
    for (int i = 0; i < 8; i++) {
      int qg = q0 + wave * 16 + hf * 8 + i;
      float v = o[nb][i] / lrow[i];
      out[((size_t)bb * S + qg) * Dm + head * HD + nb * 16 + r] = f32_to_bf16(v);
    }
  }
}

// ------------------------------- launcher -----------------------------------
extern "C" void kernel_launch(void* const* d_in, const int* in_sizes, int n_in,
                              void* d_out, int out_size, void* d_ws, size_t ws_size,
                              hipStream_t stream) {
  const int Bb = 2, S = 2048, D = 1024, H = 16, HD = 64;
  const int M = Bb * S;                 // 4096
  const int D3 = 3 * D, D4 = 4 * D;

  const float* x    = (const float*)d_in[0];
  const float* Wqkv = (const float*)d_in[1];
  const float* bqkv = (const float*)d_in[2];
  const float* Wo   = (const float*)d_in[3];
  const float* bo   = (const float*)d_in[4];
  const float* Wfc  = (const float*)d_in[5];
  const float* bfc  = (const float*)d_in[6];
  const float* Wpr  = (const float*)d_in[7];
  const float* bpr  = (const float*)d_in[8];
  const float* g1   = (const float*)d_in[9];
  const float* be1  = (const float*)d_in[10];
  const float* g2   = (const float*)d_in[11];
  const float* be2  = (const float*)d_in[12];
  float* out = (float*)d_out;

  char* ws = (char*)d_ws;
  size_t off = 0;
  auto carve = [&](size_t bytes) -> char* {
    char* p = ws + off;
    off = (off + bytes + 255) & ~(size_t)255;
    return p;
  };
  unsigned short* wq   = (unsigned short*)carve((size_t)D * D3 * 2);
  unsigned short* wo   = (unsigned short*)carve((size_t)D * D  * 2);
  unsigned short* wfc  = (unsigned short*)carve((size_t)D * D4 * 2);
  unsigned short* wpr  = (unsigned short*)carve((size_t)D4 * D * 2);
  unsigned short* ln1  = (unsigned short*)carve((size_t)M * D  * 2);
  unsigned short* qkv  = (unsigned short*)carve((size_t)M * D3 * 2);
  unsigned short* aout = (unsigned short*)carve((size_t)M * D  * 2);
  float*          x1   = (float*)         carve((size_t)M * D  * 4);
  unsigned short* ln2  = (unsigned short*)carve((size_t)M * D  * 2);
  unsigned short* h2   = (unsigned short*)carve((size_t)M * D4 * 2);

  auto cvt = [&](const float* s, unsigned short* d, int n) {
    cvt_bf16_kernel<<<(n + 255) / 256, 256, 0, stream>>>(s, d, n);
  };
  cvt(Wqkv, wq,  D * D3);
  cvt(Wo,   wo,  D * D);
  cvt(Wfc,  wfc, D * D4);
  cvt(Wpr,  wpr, D4 * D);

  // h = LN1(x)
  ln_bf16_kernel<<<M, 256, 0, stream>>>(x, g1, be1, ln1, D);

  // qkv = h @ Wqkv + b   (bf16 out)
  gemm_bf16_kernel<false, false, false, true>
      <<<dim3(M / 128, D3 / 128), 256, 0, stream>>>(
          ln1, wq, bqkv, nullptr, nullptr, qkv, M, D3, D);

  // flash attention -> aout (bf16)
  attn_kernel<<<dim3(S / 128, H, Bb), 256, 0, stream>>>(qkv, aout, S, H, HD);

  // x1 = x + aout @ Wo + bo   (fp32)
  gemm_bf16_kernel<false, true, true, false>
      <<<dim3(M / 128, D / 128), 256, 0, stream>>>(
          aout, wo, bo, x, x1, nullptr, M, D, D);

  // h2 = GELU(LN2(x1) @ Wfc + bfc)  (bf16)
  ln_bf16_kernel<<<M, 256, 0, stream>>>(x1, g2, be2, ln2, D);
  gemm_bf16_kernel<true, false, false, true>
      <<<dim3(M / 128, D4 / 128), 256, 0, stream>>>(
          ln2, wfc, bfc, nullptr, nullptr, h2, M, D4, D);

  // out = x1 + h2 @ Wpr + bpr  (fp32)
  gemm_bf16_kernel<false, true, true, false>
      <<<dim3(M / 128, D / 128), 256, 0, stream>>>(
          h2, wpr, bpr, x1, out, nullptr, M, D, D4);
}
